// ModelClass_78752520340010
// MI455X (gfx1250) — matrix operand order repaired and checked by reading, not compiled, not run on hardware
//
#include <hip/hip_runtime.h>

typedef __attribute__((ext_vector_type(16))) _Float16 v16h;
typedef __attribute__((ext_vector_type(8)))  float    v8f;

static __device__ __forceinline__ v8f wmma16(v16h a, v16h b, v8f c) {
  return __builtin_amdgcn_wmma_f32_16x16x32_f16(false, a, false, b, (short)0, c,
                                                false, false);
}

// A fragment (16x32 f16, MxK) from an LDS panel [16][ldk], row-major, K base kb.
// ISA 7.12.2: lanes 0-15 row=lane, halves K={kb+0..7, kb+16..23};
// lanes 16-31 row=lane-16, halves K={kb+8..15, kb+24..31}.
static __device__ __forceinline__ v16h frag_a_lds(const _Float16* __restrict__ A,
                                                  int ldk, int kb, int lane) {
  int row  = lane & 15;
  int koff = kb + (((lane >> 4) & 1) ? 8 : 0);
  const _Float16* p = A + row * ldk + koff;
  v16h a;
#pragma unroll
  for (int i = 0; i < 8; ++i) { a[i] = p[i]; a[i + 8] = p[i + 16]; }
  return a;
}

// Packed B fragment: layout [ntile][kchunk][lane][16] halfs, pre-zero-padded.
static __device__ __forceinline__ v16h frag_b_pk(const _Float16* __restrict__ Wp,
                                                 int nkc, int kc, int nt, int lane) {
  return *(const v16h*)(Wp + (((size_t)nt * nkc + kc) * 32 + lane) * 16);
}

static __device__ __forceinline__ float lrelu(float v) {
  return v > 0.f ? v : 0.2f * v;
}

// ---- pack W[K][N] (f32 row-major) into WMMA B-fragment layout (f16) --------
// out[i], i = ((nt*nkc + kc)*32 + lane)*16 + j ;
// element = W[kc*32 + (lane>=16?8:0) + (j&7) + (j>=8?16:0)][nt*16 + (lane&15)], OOB -> 0.
__global__ void k_pack(const float* __restrict__ W, int K, int N, int Kpad,
                       _Float16* __restrict__ out, int total) {
  int i = blockIdx.x * 256 + threadIdx.x;
  if (i >= total) return;
  int nkc  = Kpad >> 5;
  int j    = i & 15;
  int lane = (i >> 4) & 31;
  int t    = i >> 9;
  int kc   = t % nkc;
  int nt   = t / nkc;
  int k    = kc * 32 + ((lane >> 4) << 3) + (j & 7) + ((j >> 3) << 4);
  int col  = nt * 16 + (lane & 15);
  float v  = (k < K && col < N) ? W[(size_t)k * N + col] : 0.f;
  out[i] = (_Float16)v;
}

// ---------------- DynHLVs: g[B,8] = glb2(lrelu(glb1(mean(x[:, :, :3])))) ----
__global__ void k_global(const float* __restrict__ x, int N, int stride,
                         const float* __restrict__ W1, const float* __restrict__ b1,
                         const float* __restrict__ W2, const float* __restrict__ b2,
                         float* __restrict__ g) {
  int graph = blockIdx.x;
  const float* xg = x + (size_t)graph * N * stride;
  __shared__ float mean[3];
  __shared__ float h[64];
  int t = threadIdx.x;
  if (t < 3) {
    float s = 0.f;
    for (int n = 0; n < N; ++n) s += xg[(size_t)n * stride + t];
    mean[t] = s / (float)N;
  }
  __syncthreads();
  {
    float acc = b1[t];
    for (int k = 0; k < 3; ++k) acc += mean[k] * W1[k * 64 + t];
    h[t] = lrelu(acc);
  }
  __syncthreads();
  if (t < 8) {
    float acc = b2[t];
    for (int k = 0; k < 64; ++k) acc += h[k] * W2[k * 8 + t];
    g[graph * 8 + t] = acc;
  }
}

// ---------------- Branching: out[B,Nl*b,fl] = branch([x,g]) + parent --------
__global__ void k_branch(const float* __restrict__ x, const float* __restrict__ g,
                         const _Float16* __restrict__ Wp, const float* __restrict__ bias,
                         float* __restrict__ out, int Nl, int fl, int b, int R) {
  const int K    = fl + 8;
  const int Kpad = (K + 31) & ~31;     // <= 128
  const int nkc  = Kpad >> 5;
  const int Nout = b * fl;             // multiple of 16
  __shared__ _Float16 Asm[4][16 * 128];
  int wave = threadIdx.x >> 5, lane = threadIdx.x & 31;
  int row0 = (blockIdx.x * 4 + wave) * 16;
  if (row0 >= R) return;
  _Float16* Alds = Asm[wave];
  {
    int r = lane & 15, h = lane >> 4;
    int grow = row0 + r;
    int graph = grow / Nl;
    const float* xrow = x + (size_t)grow * fl;
    const float* gp   = g + graph * 8;
    for (int k = h; k < Kpad; k += 2) {
      float v = 0.f;
      if (k < fl)      v = xrow[k];
      else if (k < K)  v = gp[k - fl];
      Alds[r * Kpad + k] = (_Float16)v;
    }
  }
  int rbase = (lane >> 4) * 8, colL = lane & 15;
  for (int nt = 0; nt < Nout / 16; ++nt) {
    v8f acc = {};
    for (int kc = 0; kc < nkc; ++kc) {
      v16h a  = frag_a_lds(Alds, Kpad, kc * 32, lane);
      v16h bw = frag_b_pk(Wp, nkc, kc, nt, lane);
      acc = wmma16(a, bw, acc);
    }
    int col = nt * 16 + colL;
    int j = col / fl, f = col - j * fl;
    float bb = bias[col];
#pragma unroll
    for (int i = 0; i < 8; ++i) {
      int grow = row0 + rbase + i;          // graph*Nl+node == grow
      out[((size_t)grow * b + j) * fl + f] = acc[i] + bb + x[(size_t)grow * fl + f];
    }
  }
}

// ---------------- Ancestor conv: out (+)= anc2(lrelu(anc1([src,ea,g]))) ----
__global__ void k_anc(const float* __restrict__ src, int srcN, int srcW, int rep,
                      float ea, const float* __restrict__ g,
                      const _Float16* __restrict__ W1p, const float* __restrict__ b1,
                      const _Float16* __restrict__ W2p, const float* __restrict__ b2,
                      float* __restrict__ out, int Nn, int fl, int fn, int R,
                      int accumulate) {
  const int K1    = fl + 9;
  const int K1pad = (K1 + 31) & ~31;   // <= 128
  const int nkc1  = K1pad >> 5;
  __shared__ _Float16 Asm[4][16 * 128];
  __shared__ _Float16 Hsm[4][16 * 64];
  int wave = threadIdx.x >> 5, lane = threadIdx.x & 31;
  int row0 = (blockIdx.x * 4 + wave) * 16;
  if (row0 >= R) return;
  _Float16* Alds = Asm[wave];
  _Float16* Hlds = Hsm[wave];
  {
    int r = lane & 15, h = lane >> 4;
    int grow = row0 + r;
    int graph = grow / Nn, node = grow - graph * Nn;
    const float* srow = src + ((size_t)graph * srcN + node / rep) * srcW;
    const float* gp   = g + graph * 8;
    for (int k = h; k < K1pad; k += 2) {
      float v = 0.f;
      if (k < fl)       v = srow[k];
      else if (k == fl) v = ea;
      else if (k < K1)  v = gp[k - fl - 1];
      Alds[r * K1pad + k] = (_Float16)v;
    }
  }
  int rbase = (lane >> 4) * 8, colL = lane & 15;
  // GEMM1: hidden 64 + bias + lrelu -> LDS f16
  for (int ht = 0; ht < 4; ++ht) {
    v8f acc = {};
    for (int kc = 0; kc < nkc1; ++kc) {
      v16h a  = frag_a_lds(Alds, K1pad, kc * 32, lane);
      v16h bw = frag_b_pk(W1p, nkc1, kc, ht, lane);
      acc = wmma16(a, bw, acc);
    }
    int col = ht * 16 + colL;
    float bb = b1[col];
#pragma unroll
    for (int i = 0; i < 8; ++i)
      Hlds[(rbase + i) * 64 + col] = (_Float16)lrelu(acc[i] + bb);
  }
  // GEMM2: K=64 -> fn, accumulate into out
  for (int nt = 0; nt < (fn + 15) / 16; ++nt) {
    v8f acc = {};
#pragma unroll
    for (int kc = 0; kc < 2; ++kc) {
      v16h a  = frag_a_lds(Hlds, 64, kc * 32, lane);
      v16h bw = frag_b_pk(W2p, 2, kc, nt, lane);
      acc = wmma16(a, bw, acc);
    }
    int col = nt * 16 + colL;
    if (col < fn) {
      float bb = b2[col];
#pragma unroll
      for (int i = 0; i < 8; ++i) {
        size_t o = (size_t)(row0 + rbase + i) * fn + col;
        float v = acc[i] + bb;
        out[o] = accumulate ? (out[o] + v) : v;
      }
    }
  }
}

// ---------------- MPLSeq: two fused sibling message-passing layers ----------
__global__ void k_mpl(const float* __restrict__ xin, const float* __restrict__ g,
                      const _Float16* __restrict__ W1a, const float* __restrict__ b1a,
                      const _Float16* __restrict__ W2a, const float* __restrict__ b2a,
                      const _Float16* __restrict__ W1b, const float* __restrict__ b1b,
                      const _Float16* __restrict__ W2b, const float* __restrict__ b2b,
                      float* __restrict__ out, int Nn, int fn, int b, int R) {
  __shared__ _Float16 Xsm[4][16 * 64];
  __shared__ _Float16 Gsm[4][16 * 8];
  __shared__ _Float16 Asm[4][16 * 160];
  __shared__ float    Msm[4][16 * 64];
  int wave = threadIdx.x >> 5, lane = threadIdx.x & 31;
  int row0 = (blockIdx.x * 4 + wave) * 16;
  if (row0 >= R) return;
  _Float16* Xs = Xsm[wave];
  _Float16* Gs = Gsm[wave];
  _Float16* As = Asm[wave];
  float*    Ms = Msm[wave];

  const int K1    = fn + 8;
  const int K1pad = (K1 + 31) & ~31;     // <= 96
  const int nkc1  = K1pad >> 5;
  const int K2    = fn + 64 + 8;
  const int K2pad = (K2 + 31) & ~31;     // <= 160
  const int nkc2  = K2pad >> 5;
  int r = lane & 15, h = lane >> 4;
  int rbase = h * 8, colL = r;
  int grow = row0 + r, graph = grow / Nn;
  int lb = (r / b) * b;                   // sibling-group base (tiles b-aligned)

  for (int k = h; k < fn; k += 2)
    Xs[r * fn + k] = (_Float16)xin[(size_t)grow * fn + k];
  if (h == 0) {
    const float* gp = g + graph * 8;
    for (int c = 0; c < 8; ++c) Gs[r * 8 + c] = (_Float16)gp[c];
  }

  for (int layer = 0; layer < 2; ++layer) {
    const _Float16* W1 = layer ? W1b : W1a;  const float* B1 = layer ? b1b : b1a;
    const _Float16* W2 = layer ? W2b : W2a;  const float* B2 = layer ? b2b : b2a;
    // A = [x, g]
    for (int k = h; k < K1pad; k += 2) {
      _Float16 v = (_Float16)0.f;
      if (k < fn)      v = Xs[r * fn + k];
      else if (k < K1) v = Gs[r * 8 + (k - fn)];
      As[r * K1pad + k] = v;
    }
    // msg GEMM -> Ms (f32, post-lrelu)
    for (int ht = 0; ht < 4; ++ht) {
      v8f acc = {};
      for (int kc = 0; kc < nkc1; ++kc) {
        v16h a  = frag_a_lds(As, K1pad, kc * 32, lane);
        v16h bw = frag_b_pk(W1, nkc1, kc, ht, lane);
        acc = wmma16(a, bw, acc);
      }
      int col = ht * 16 + colL;
      float bb = B1[col];
#pragma unroll
      for (int i = 0; i < 8; ++i)
        Ms[(rbase + i) * 64 + col] = lrelu(acc[i] + bb);
    }
    // A2 = [x, agg (= sibling sum - self), g]
    for (int k = h; k < K2pad; k += 2) {
      float v = 0.f;
      if (k < fn) v = (float)Xs[r * fn + k];
      else if (k < fn + 64) {
        int c = k - fn;
        float s = 0.f;
        for (int j = 0; j < b; ++j) s += Ms[(lb + j) * 64 + c];
        v = s - Ms[r * 64 + c];
      } else if (k < K2) v = (float)Gs[r * 8 + (k - fn - 64)];
      As[r * K2pad + k] = (_Float16)v;
    }
    // upd GEMM -> lrelu -> Xs (layer 0) or global out (layer 1)
    for (int nt = 0; nt < (fn + 15) / 16; ++nt) {
      v8f acc = {};
      for (int kc = 0; kc < nkc2; ++kc) {
        v16h a  = frag_a_lds(As, K2pad, kc * 32, lane);
        v16h bw = frag_b_pk(W2, nkc2, kc, nt, lane);
        acc = wmma16(a, bw, acc);
      }
      int col = nt * 16 + colL;
      if (col < fn) {
        float bb = B2[col];
#pragma unroll
        for (int i = 0; i < 8; ++i) {
          float v = lrelu(acc[i] + bb);
          if (layer == 0) Xs[(rbase + i) * fn + col] = (_Float16)v;
          else            out[(size_t)(row0 + rbase + i) * fn + col] = v;
        }
      }
    }
  }
}

// ---------------------------------------------------------------------------
extern "C" void kernel_launch(void* const* d_in, const int* in_sizes, int n_in,
                              void* d_out, int out_size, void* d_ws, size_t ws_size,
                              hipStream_t stream) {
  (void)in_sizes; (void)n_in; (void)out_size; (void)ws_size;
  const int FE[5] = {96, 64, 32, 16, 3};
  const int BR[4] = {2, 8, 16, 16};
  const int NL[4] = {1, 2, 16, 256};
  const int NN[4] = {2, 16, 256, 4096};
  const int B = 128;

  const float* rv = (const float*)d_in[0];
  auto P = [&](int l, int j) -> const float* {
    return (const float*)d_in[1 + l * 18 + j];
  };
  auto c32 = [](int v) { return (v + 31) & ~31; };
  auto c16 = [](int v) { return (v + 15) & ~15; };

  // ---- pack all weights into WMMA B-fragment layout at the head of ws ----
  // matrices per level: branch, anc1, anc2, msg0, upd0, msg1, upd1
  _Float16* pkcur = (_Float16*)d_ws;
  _Float16* pk[4][7];
  for (int l = 0; l < 4; ++l) {
    int fl = FE[l], fn = FE[l + 1], b = BR[l];
    const int pidx[7] = {4, 6, 8, 10, 12, 14, 16};
    const int Ks[7]   = {fl + 8, fl + 9, 64, fn + 8, fn + 72, fn + 8, fn + 72};
    const int Ns[7]   = {b * fl, 64, fn, 64, fn, 64, fn};
    for (int m = 0; m < 7; ++m) {
      int Kp = c32(Ks[m]), Np = c16(Ns[m]);
      int total = Kp * Np;
      k_pack<<<(total + 255) / 256, 256, 0, stream>>>(P(l, pidx[m]), Ks[m], Ns[m],
                                                      Kp, pkcur, total);
      pk[l][m] = pkcur;
      pkcur += total;
    }
  }
  size_t pkbytes = ((char*)pkcur - (char*)d_ws + 255) & ~(size_t)255;
  float* fbase = (float*)((char*)d_ws + pkbytes);

  float* gbf = fbase;               // B*8           = 1024
  float* x1  = gbf + 1024;          // B*2*64        = 16384
  float* x2  = x1 + 16384;          // B*16*32       = 65536
  float* x3  = x2 + 65536;          // B*256*16      = 524288
  float* xb  = x3 + 524288;         // B*4096*16 max = 8388608
  float* xc  = xb + 8388608;        // B*4096*3  max = 1572864
  float* outp = (float*)d_out;

  float* lvl_ptr[5]   = {(float*)rv, x1, x2, x3, outp};
  int    lvl_nodes[5] = {1, 2, 16, 256, 4096};

  for (int l = 0; l < 4; ++l) {
    int fl = FE[l], fn = FE[l + 1], b = BR[l], Nl = NL[l], Nn = NN[l];
    const float* x = lvl_ptr[l];

    // 1) per-graph globals
    k_global<<<B, 64, 0, stream>>>(x, Nl, fl, P(l, 0), P(l, 1), P(l, 2), P(l, 3), gbf);

    // 2) branching (+parent residual)
    int Rb = B * Nl;
    k_branch<<<(Rb / 16 + 3) / 4, 128, 0, stream>>>(x, gbf, pk[l][0], P(l, 5), xb,
                                                    Nl, fl, b, Rb);

    // 3) ancestor conv: one pass per ancestor source, accumulate into xc
    int R = B * Nn;
    dim3 gridA((R / 16 + 3) / 4);
    for (int a = 0; a <= l + 1; ++a) {
      const float* src; int srcN, srcW;
      if (a <= l) { src = lvl_ptr[a]; srcN = lvl_nodes[a]; srcW = FE[a]; }
      else        { src = xb;         srcN = Nn;           srcW = fl;    }
      int rep  = Nn / srcN;
      float ea = (float)(l + 1 - a);
      k_anc<<<gridA, 128, 0, stream>>>(src, srcN, srcW, rep, ea, gbf,
                                       pk[l][1], P(l, 7), pk[l][2], P(l, 9),
                                       xc, Nn, fl, fn, R, a == 0 ? 0 : 1);
    }

    // 4) fused 2-layer sibling message passing
    float* xout = lvl_ptr[l + 1];
    k_mpl<<<gridA, 128, 0, stream>>>(xc, gbf,
                                     pk[l][3], P(l, 11), pk[l][4], P(l, 13),
                                     pk[l][5], P(l, 15), pk[l][6], P(l, 17),
                                     xout, Nn, fn, b, R);
  }
}
